// CGATransformPipeline_90477781057899
// MI455X (gfx1250) — matrix operand order repaired and checked by reading, not compiled, not run on hardware
//
#include <hip/hip_runtime.h>
#include <stdint.h>

typedef _Float16 v16h __attribute__((ext_vector_type(16)));
typedef uint32_t v8u  __attribute__((ext_vector_type(8)));
typedef float    v8f  __attribute__((ext_vector_type(8)));

// ---------------------------------------------------------------------------
// Compile-time construction of the Cl(4,1) sandwich-product tensor, with the
// UPGC encode (p = [x1,x2,x3, s-0.5, s+0.5], s=0.5|x|^2) and decode
// (out = t[:3]/(t4-t3)) affine maps folded in, then pre-swizzled into the
// exact per-lane VGPR layout of the WMMA_F32_16X16X32_F16 A-operand.
// ---------------------------------------------------------------------------
constexpr int MB[16] = {0b00000,0b00011,0b00101,0b01001,0b10001,0b00110,0b01010,0b10010,
                        0b01100,0b10100,0b11000,0b01111,0b10111,0b11011,0b11101,0b11110};
constexpr int PB[5]  = {0b00001,0b00010,0b00100,0b01000,0b10000};

constexpr int popc5(int x){ int c=0; for(int b=0;b<8;b++) c += (x>>b)&1; return c; }
constexpr int gp_sign(int a,int b){
  int s=0; int aa=a>>1;
  while(aa){ s += popc5(aa&b); aa>>=1; }
  int sign = (s&1)? -1 : 1;
  int common = a&b;
  for(int bit=0;bit<5;bit++) if((common>>bit)&1) sign *= (bit==4? -1 : 1);
  return sign;
}
constexpr int rev_sign(int blade){ int g=popc5(blade); return ((g*(g-1)/2)&1)? -1 : 1; }

struct Cfold { float v[20][16][16]; };   // row = o'*5 + j', cols = (i,k) motor pair
constexpr Cfold build_cfold(){
  float C[5][16][5][16] = {};
  for(int i=0;i<16;i++) for(int j=0;j<5;j++){
    int s1 = gp_sign(MB[i], PB[j]);
    int ab = MB[i] ^ PB[j];
    for(int k=0;k<16;k++){
      int out = ab ^ MB[k];
      for(int o=0;o<5;o++) if(out==PB[o])
        C[o][i][j][k] += (float)(s1 * gp_sign(ab, MB[k]) * rev_sign(MB[k]));
    }
  }
  Cfold f = {};
  for(int op=0; op<4; op++) for(int i=0;i<16;i++) for(int k=0;k<16;k++){
    float R[5];
    for(int j=0;j<5;j++)
      R[j] = (op<3)? C[op][i][j][k] : (C[4][i][j][k]-C[3][i][j][k]);  // row3 = t4-t3 (scale)
    f.v[op*5+0][i][k] = R[0];                 // * x1
    f.v[op*5+1][i][k] = R[1];                 // * x2
    f.v[op*5+2][i][k] = R[2];                 // * x3
    f.v[op*5+3][i][k] = R[3]+R[4];            // * s
    f.v[op*5+4][i][k] = 0.5f*(R[4]-R[3]);     // * 1
  }
  return f;
}

// Pre-swizzled A-operand: [lane][Mtile][Kchunk][16 halves].
// A layout (16x32), contiguous K order to match B: lane holds row M=lane%16; lanes<16: K=0..15,
// lanes>=16: K=16..31. Half slot s -> K=s(+16 hi).
struct Swz { alignas(32) _Float16 v[32][2][8][16]; };
constexpr Swz build_swz(){
  Cfold cf = build_cfold();
  Swz s = {};
  for(int lane=0;lane<32;lane++){
    int hi = lane>>4, col = lane&15;
    for(int t=0;t<2;t++) for(int c=0;c<8;c++) for(int e=0;e<16;e++){
      int row = t*16 + col;
      int kl  = e + hi*16;
      int K   = c*32 + kl;                 // K = i*16 + k
      float val = (row<20)? cf.v[row][K>>4][K&15] : 0.0f;
      s.v[lane][t][c][e] = (_Float16)val;
    }
  }
  return s;
}
__device__ constexpr Swz gSwz = build_swz();

// ---------------------------------------------------------------------------
// Per wave: 16 points (one per D column). B-operand chunk c for lane L is
// m[2c + (L>=16)] * m[0..15] of point L%16  (B layout: lanes<16 hold K=0..15,
// lanes>=16 hold K=16..31 of each 32-K chunk). 16 WMMA per 16 points.
// ---------------------------------------------------------------------------
__global__ void __launch_bounds__(256)
cga_motor_sandwich(const _Float16* __restrict__ motor,
                   const _Float16* __restrict__ x,
                   _Float16* __restrict__ out,
                   int npts, int ntiles, int nwaves)
{
  const int tid  = blockIdx.x*blockDim.x + threadIdx.x;
  const int wave = tid >> 5;
  const int lane = threadIdx.x & 31;
  const int hi   = lane >> 4;
  const int col  = lane & 15;
  const uint32_t shamt = hi ? 16u : 0u;    // per-lane half-select, computed once

  // One-time: pull this lane's slice of the constant A-operand (coalesced b128s),
  // held in VGPRs across the whole grid-stride loop.
  v16h aT[2][8];
  {
    const v16h* tp = (const v16h*)(&gSwz.v[lane][0][0][0]);
    #pragma unroll
    for(int t=0;t<2;t++)
      #pragma unroll
      for(int c=0;c<8;c++) aT[t][c] = tp[t*8+c];
  }

  for(int tile = wave; tile < ntiles; tile += nwaves){
    const int P  = tile*16 + col;
    const int Pc = (P < npts) ? P : (npts-1);

    const v16h m  = *(const v16h*)(motor + (size_t)Pc*16);   // 32B, 32B-aligned
    const v8u  mu = __builtin_bit_cast(v8u, m);              // dword c = {m[2c], m[2c+1]}

    v8f acc0 = {}, acc1 = {};
    #pragma unroll
    for(int c=0;c<8;c++){
      // scalar m[2c + hi] via one v_lshrrev_b32 (no dynamic-index cndmask trees)
      const uint16_t scBits = (uint16_t)(mu[c] >> shamt);
      const _Float16 sc = __builtin_bit_cast(_Float16, scBits);
      const v16h b = m * sc;                                 // 8x v_pk_mul_f16 (op_sel broadcast)
      acc0 = __builtin_amdgcn_wmma_f32_16x16x32_f16(false, aT[0][c], false, b,
                                                    (short)0, acc0, false, false);
      acc1 = __builtin_amdgcn_wmma_f32_16x16x32_f16(false, aT[1][c], false, b,
                                                    (short)0, acc1, false, false);
    }

    // p' = [x1, x2, x3, s, 1] in f32 (matches reference fp32 path)
    const _Float16* xp = x + (size_t)Pc*3;
    const float X0=(float)xp[0], X1=(float)xp[1], X2=(float)xp[2];
    const float p3 = 0.5f*(X0*X0 + X1*X1 + X2*X2);

    // D column = point; lanes<16 hold rows M=0..7 (+tile1 16..19), lanes>=16 rows 8..15.
    float u0=0.f, u1=0.f, u2=0.f, u3=0.f;
    if(!hi){
      u0 = acc0[0]*X0 + acc0[1]*X1 + acc0[2]*X2 + acc0[3]*p3 + acc0[4];
      u1 = acc0[5]*X0 + acc0[6]*X1 + acc0[7]*X2;
      u3 = acc1[0]*X1 + acc1[1]*X2 + acc1[2]*p3 + acc1[3];          // rows 16..19 = (3,1..4)
    } else {
      u1 = acc0[0]*p3 + acc0[1];                                     // rows 8,9 = (1,3..4)
      u2 = acc0[2]*X0 + acc0[3]*X1 + acc0[4]*X2 + acc0[5]*p3 + acc0[6];
      u3 = acc0[7]*X0;                                               // row 15 = (3,0)
    }
    u0 += __shfl_xor(u0, 16, 32);
    u1 += __shfl_xor(u1, 16, 32);
    u2 += __shfl_xor(u2, 16, 32);
    u3 += __shfl_xor(u3, 16, 32);

    if(!hi && P < npts){
      const float inv = 1.0f / u3;          // u3 = t4 - t3 (homogeneous scale)
      _Float16* op = out + (size_t)P*3;
      op[0] = (_Float16)(u0*inv);
      op[1] = (_Float16)(u1*inv);
      op[2] = (_Float16)(u2*inv);
    }
  }
}

extern "C" void kernel_launch(void* const* d_in, const int* in_sizes, int n_in,
                              void* d_out, int out_size, void* d_ws, size_t ws_size,
                              hipStream_t stream)
{
  (void)n_in; (void)out_size; (void)d_ws; (void)ws_size;
  const _Float16* motor = (const _Float16*)d_in[0];
  const _Float16* xin   = (const _Float16*)d_in[1];
  _Float16* out = (_Float16*)d_out;

  const int npts   = in_sizes[0] / 16;
  const int ntiles = (npts + 15) / 16;

  const int threads = 256;                 // 8 waves per workgroup (wave32)
  int blocks = 1024;                       // 8192 waves -> ~16 tiles/wave, amortizes const load
  const int maxBlocks = (ntiles + (threads/32) - 1) / (threads/32);
  if (blocks > maxBlocks) blocks = maxBlocks > 0 ? maxBlocks : 1;
  const int nwaves = blocks * (threads/32);

  cga_motor_sandwich<<<blocks, threads, 0, stream>>>(motor, xin, out, npts, ntiles, nwaves);
}
